// PointNetKnnInterpolator_78005196030106
// MI455X (gfx1250) — matrix operand order repaired and checked by reading, not compiled, not run on hardware
//
#include <hip/hip_runtime.h>

typedef __attribute__((ext_vector_type(16))) _Float16 v16h;
typedef __attribute__((ext_vector_type(8)))  _Float16 v8h;
typedef __attribute__((ext_vector_type(4)))  _Float16 v4h;
typedef __attribute__((ext_vector_type(8)))  float    v8f;
typedef _Float16 half_t;

#define TILE_M   16        // edges per wave-tile (= 2 output points, K_NN = 8)
#define WAVES_PB 4         // waves per block (128 threads)
#define HSTRIDE  96        // activation row stride in halves (16B-aligned, K padded)

// Transposed f16 weight arena (units: half_t): each matrix stored [n][Kpad].
// W0a/Wsa padded to Kpad=96 (k 67..95 = 0); W1a/W0b/W1b have Kpad=64.
#define W0A_OFF  0
#define WSA_OFF  (64*96)
#define W1A_OFF  (2*64*96)
#define W0B_OFF  (2*64*96 + 64*64)
#define W1B_OFF  (2*64*96 + 2*64*64)
#define WTOT     (2*64*96 + 3*64*64)   // 24576 halves = 49152 bytes

static __device__ __forceinline__ v8f wmma_f16(v16h a, v16h b, v8f c) {
  // (neg_a, A, neg_b, B, c_mod, C, reuse_a, reuse_b)
  return __builtin_amdgcn_wmma_f32_16x16x32_f16(false, a, false, b, (short)0, c,
                                                false, false);
}

// A fragment (16x32 tile, row-major activation buffer, 16B-aligned rows).
// half e -> k = kbase + e + 8*((e>=8) + hi): two contiguous 8-half runs.
static __device__ __forceinline__ v16h load_a_frag(const half_t* act,
                                                   int row_off, int kbase,
                                                   int hi) {
  const half_t* p = act + row_off + kbase + 8 * hi;
  const v8h lo = *(const v8h*)(p);
  const v8h hh = *(const v8h*)(p + 16);
  v16h a;
#pragma unroll
  for (int e = 0; e < 8; ++e) { a[e] = lo[e]; a[e + 8] = hh[e]; }
  return a;
}

// B fragment (32x16 weight tile) from transposed [n][Kpad] storage.
// half e -> k = kbase + e + 8*((e>=8) + hi): two contiguous 8-half runs.
static __device__ __forceinline__ v16h load_b_frag(const half_t* wt, int n,
                                                   int Kpad, int kbase,
                                                   int hi) {
  const half_t* p = wt + n * Kpad + kbase + 8 * hi;
  const v8h lo = *(const v8h*)(p);
  const v8h hh = *(const v8h*)(p + 16);
  v16h b;
#pragma unroll
  for (int e = 0; e < 8; ++e) { b[e] = lo[e]; b[e + 8] = hh[e]; }
  return b;
}

static __device__ __forceinline__ v16h relu16(v16h a) {
  v16h r;
#pragma unroll
  for (int e = 0; e < 16; ++e) r[e] = (a[e] > (half_t)0.f) ? a[e] : (half_t)0.f;
  return r;
}

// ---------------------------------------------------------------------------
// Prep: fp32 weights -> transposed, K-padded f16 arena in d_ws.
// ---------------------------------------------------------------------------
__global__ void prep_weights_kernel(const float* __restrict__ W0a,
                                    const float* __restrict__ Wsa,
                                    const float* __restrict__ W1a,
                                    const float* __restrict__ W0b,
                                    const float* __restrict__ W1b,
                                    half_t* __restrict__ ws) {
  const int i = blockIdx.x * blockDim.x + threadIdx.x;
  if (i >= WTOT) return;
  float v = 0.f;
  if (i < WSA_OFF) {
    const int n = i / 96, k = i % 96;
    if (k < 67) v = W0a[k * 64 + n];
  } else if (i < W1A_OFF) {
    const int j = i - WSA_OFF, n = j / 96, k = j % 96;
    if (k < 67) v = Wsa[k * 64 + n];
  } else if (i < W0B_OFF) {
    const int j = i - W1A_OFF, n = j >> 6, k = j & 63;
    v = W1a[k * 64 + n];
  } else if (i < W1B_OFF) {
    const int j = i - W0B_OFF, n = j >> 6, k = j & 63;
    v = W0b[k * 64 + n];
  } else {
    const int j = i - W1B_OFF, n = j >> 6, k = j & 63;
    v = W1b[k * 64 + n];
  }
  ws[i] = (half_t)v;
}

// ---------------------------------------------------------------------------
// Main: one wave per 16-edge tile; 48 v_wmma_f32_16x16x32_f16 per tile;
// per-lane segment-max over 8 rows -> 2 output points per tile, no atomics.
// ---------------------------------------------------------------------------
__global__ void __launch_bounds__(32 * WAVES_PB)
pointnet_tile_kernel(const float* __restrict__ x,
                     const float* __restrict__ pos_x,
                     const float* __restrict__ pos_y,
                     const int*   __restrict__ x_idx,
                     const float* __restrict__ b0a,
                     const float* __restrict__ b1a,
                     const float* __restrict__ b0b,
                     const float* __restrict__ b1b,
                     const half_t* __restrict__ wsrc,
                     float* __restrict__ out, int ntiles) {
  __shared__ __align__(16) half_t sW[WTOT];                      // 49152 B
  __shared__ __align__(16) half_t sAct[WAVES_PB][TILE_M * HSTRIDE]; // 12288 B

  const int tid = threadIdx.x;

  // Cooperative weight stage: b128 global loads -> b128 LDS stores.
  {
    const uint4* src = (const uint4*)wsrc;
    uint4* dst = (uint4*)sW;
    for (int i = tid; i < WTOT / 8; i += blockDim.x) dst[i] = src[i];
  }
  __syncthreads();

  const int wv   = tid >> 5;
  const int lane = tid & 31;
  const int tile = blockIdx.x * WAVES_PB + wv;   // uniform per wave
  if (tile >= ntiles) return;

  half_t* AB = &sAct[wv][0];      // staged h, then reused as layer scratch
  const int m  = lane & 15;       // row (edge within tile) / local out column
  const int hi = lane >> 4;       // lane-half select
  const int row_off = m * HSTRIDE;
  const int e0 = tile * TILE_M;

  // ---- Stage h = [x[x_idx[e]], diff, 0-pad] as f16, packed stores ----
  {
    const int e   = e0 + m;
    const int idx = x_idx[e];
    const float4* xr4 = (const float4*)(x + (size_t)idx * 64) + hi * 8;
    half_t* hrow = AB + row_off + hi * 32;
#pragma unroll
    for (int j = 0; j < 8; ++j) {
      const float4 v = xr4[j];
      const v4h p = {(half_t)v.x, (half_t)v.y, (half_t)v.z, (half_t)v.w};
      *(v4h*)(hrow + j * 4) = p;     // ds_store_b64
    }
    if (hi) {                        // lanes 16..31 fill diff + zero pad 64..95
      const int y = e >> 3;          // y_idx[e] = e / K_NN
      const float d0 = pos_x[idx * 3 + 0] - pos_y[y * 3 + 0];
      const float d1 = pos_x[idx * 3 + 1] - pos_y[y * 3 + 1];
      const float d2 = pos_x[idx * 3 + 2] - pos_y[y * 3 + 2];
      half_t* prow = AB + row_off + 64;
      const v8h z = {};
      v8h first = z;
      first[0] = (half_t)d0; first[1] = (half_t)d1; first[2] = (half_t)d2;
      *(v8h*)(prow +  0) = first;    // ds_store_b128 x4
      *(v8h*)(prow +  8) = z;
      *(v8h*)(prow + 16) = z;
      *(v8h*)(prow + 24) = z;
    }
  }

  // ---- A fragments of h (K padded to 96): raw for Wsa, relu'd for W0a ----
  v16h Araw[3], Arelu[3];
#pragma unroll
  for (int c = 0; c < 3; ++c) {
    Araw[c]  = load_a_frag(AB, row_off, 32 * c, hi);
    Arelu[c] = relu16(Araw[c]);
  }

  const int nl = m;  // local output column within a 16-wide N tile

  // ---- Layer 1: net = relu(h) @ W0a + b0a ; AB <- relu(net) ----
#pragma unroll
  for (int t = 0; t < 4; ++t) {
    const int n = 16 * t + nl;
    const float bv = b0a[n];
    v8f acc = {bv, bv, bv, bv, bv, bv, bv, bv};
#pragma unroll
    for (int c = 0; c < 3; ++c)
      acc = wmma_f16(Arelu[c], load_b_frag(sW + W0A_OFF, n, 96, 32 * c, hi), acc);
#pragma unroll
    for (int r = 0; r < 8; ++r)
      AB[(8 * hi + r) * HSTRIDE + n] = (half_t)fmaxf(acc[r], 0.f);
  }

  // ---- Layer 2: h2 = h @ Wsa + relu(net) @ W1a + b1a ; AB <- relu(h2) ----
  v16h Anet[2];
#pragma unroll
  for (int c = 0; c < 2; ++c) Anet[c] = load_a_frag(AB, row_off, 32 * c, hi);

  v8f h2acc[4];
#pragma unroll
  for (int t = 0; t < 4; ++t) {
    const int n = 16 * t + nl;
    const float bv = b1a[n];
    v8f acc = {bv, bv, bv, bv, bv, bv, bv, bv};
#pragma unroll
    for (int c = 0; c < 2; ++c)
      acc = wmma_f16(Anet[c], load_b_frag(sW + W1A_OFF, n, 64, 32 * c, hi), acc);
#pragma unroll
    for (int c = 0; c < 3; ++c)
      acc = wmma_f16(Araw[c], load_b_frag(sW + WSA_OFF, n, 96, 32 * c, hi), acc);
    h2acc[t] = acc;
#pragma unroll
    for (int r = 0; r < 8; ++r)
      AB[(8 * hi + r) * HSTRIDE + n] = (half_t)fmaxf(acc[r], 0.f);
  }

  // ---- Layer 3: net2 = relu(h2) @ W0b + b0b ; AB <- relu(net2) ----
  v16h Ah2[2];
#pragma unroll
  for (int c = 0; c < 2; ++c) Ah2[c] = load_a_frag(AB, row_off, 32 * c, hi);
#pragma unroll
  for (int t = 0; t < 4; ++t) {
    const int n = 16 * t + nl;
    const float bv = b0b[n];
    v8f acc = {bv, bv, bv, bv, bv, bv, bv, bv};
#pragma unroll
    for (int c = 0; c < 2; ++c)
      acc = wmma_f16(Ah2[c], load_b_frag(sW + W0B_OFF, n, 64, 32 * c, hi), acc);
#pragma unroll
    for (int r = 0; r < 8; ++r)
      AB[(8 * hi + r) * HSTRIDE + n] = (half_t)fmaxf(acc[r], 0.f);
  }

  // ---- Layer 4 + segment max: h3 = h2 + relu(net2) @ W1b + b1b ----
  v16h An2[2];
#pragma unroll
  for (int c = 0; c < 2; ++c) An2[c] = load_a_frag(AB, row_off, 32 * c, hi);
#pragma unroll
  for (int t = 0; t < 4; ++t) {
    const int n = 16 * t + nl;
    const float bv = b1b[n];
    v8f acc = h2acc[t];
#pragma unroll
    for (int r = 0; r < 8; ++r) acc[r] += bv;
#pragma unroll
    for (int c = 0; c < 2; ++c)
      acc = wmma_f16(An2[c], load_b_frag(sW + W1B_OFF, n, 64, 32 * c, hi), acc);
    // lanes 0-15 hold rows 0-7 (point y0 = 2*tile), lanes 16-31 rows 8-15 (y1)
    float mx = acc[0];
#pragma unroll
    for (int r = 1; r < 8; ++r) mx = fmaxf(mx, acc[r]);
    out[(size_t)(tile * 2 + hi) * 64 + n] = mx;
  }
}

extern "C" void kernel_launch(void* const* d_in, const int* in_sizes, int n_in,
                              void* d_out, int out_size, void* d_ws, size_t ws_size,
                              hipStream_t stream) {
  const float* x     = (const float*)d_in[0];
  const float* pos_x = (const float*)d_in[1];
  const float* pos_y = (const float*)d_in[2];
  const int*   x_idx = (const int*)d_in[3];
  // d_in[4] = y_idx (implicit: e / 8)
  const float* W0a = (const float*)d_in[5];
  const float* b0a = (const float*)d_in[6];
  const float* W1a = (const float*)d_in[7];
  const float* b1a = (const float*)d_in[8];
  const float* Wsa = (const float*)d_in[9];
  const float* W0b = (const float*)d_in[10];
  const float* b0b = (const float*)d_in[11];
  const float* W1b = (const float*)d_in[12];
  const float* b1b = (const float*)d_in[13];

  half_t* wf16 = (half_t*)d_ws;       // 49152 B transposed f16 weight arena
  float*  out  = (float*)d_out;

  prep_weights_kernel<<<(WTOT + 255) / 256, 256, 0, stream>>>(W0a, Wsa, W1a,
                                                              W0b, W1b, wf16);

  const int E = in_sizes[3];          // NY * K edges
  const int ntiles = E / TILE_M;      // 25000 tiles, 2 output points each
  dim3 grid((ntiles + WAVES_PB - 1) / WAVES_PB);
  dim3 block(32 * WAVES_PB);
  pointnet_tile_kernel<<<grid, block, 0, stream>>>(x, pos_x, pos_y, x_idx,
                                                   b0a, b1a, b0b, b1b,
                                                   wf16, out, ntiles);
}
